// GridLSTM_6012954214642
// MI455X (gfx1250) — compile-verified, matrix operand order checked by reading, not run on hardware
//
#include <hip/hip_runtime.h>
#include <hip/hip_bf16.h>
#include <math.h>

// ---------------------------------------------------------------------------
// Grid-LSTM (B=32, P=H=24, E=300, D=512) for gfx1250 using bf16 WMMA.
// Wavefront (anti-diagonal) schedule: 47 steps, 2 kernels per step.
// ---------------------------------------------------------------------------

typedef __attribute__((ext_vector_type(16))) __bf16 v16bf;
typedef __attribute__((ext_vector_type(8)))  float  v8f;
typedef __attribute__((ext_vector_type(4)))  unsigned int v4u;

union FragBF { v16bf v; v4u q[2]; };

static constexpr int Bb    = 32;     // batch
static constexpr int Pp    = 24;     // premise len
static constexpr int HhLen = 24;     // hypothesis len
static constexpr int Ee    = 300;    // embed dim
static constexpr int Dd    = 512;    // hidden dim
static constexpr int TWO_D = 1024;
static constexpr int G8D   = 4096;   // 8*D gates

// ---- fragment loader: 16-bit A/B-style fragment from row-major bf16 matrix.
// Per CDNA5 ISA 16-bit 16x32 A layout:
//   lanes 0-15  (hi=0): halves 0..7 = K0..7,  halves 8..15 = K16..23
//   lanes 16-31 (hi=1): halves 0..7 = K8..15, halves 8..15 = K24..31
__device__ __forceinline__ v16bf load_frag(const __bf16* __restrict__ base,
                                           int row, int lda, int k0, int hi) {
  FragBF f;
  const __bf16* p = base + (size_t)row * lda + k0 + hi * 8;
  f.q[0] = *reinterpret_cast<const v4u*>(p);
  f.q[1] = *reinterpret_cast<const v4u*>(p + 16);
  return f.v;
}

__device__ __forceinline__ float sigm(float x) {
  return 1.0f / (1.0f + __expf(-x));
}

// ---------------------------------------------------------------------------
// Prep: convert weights to bf16, fuse gate biases.
// ---------------------------------------------------------------------------
__global__ void prep_kernel(const float* __restrict__ Whh,
                            const float* __restrict__ Wch,
                            const float* __restrict__ Wcc,
                            const float* __restrict__ bih,
                            const float* __restrict__ bhh,
                            __bf16* __restrict__ WhhBf,
                            __bf16* __restrict__ WchBf,
                            __bf16* __restrict__ WccBf,
                            float* __restrict__ gbias) {
  size_t idx = (size_t)blockIdx.x * blockDim.x + threadIdx.x;
  const size_t n1 = (size_t)G8D * TWO_D;           // W_hh
  const size_t n2 = n1 + (size_t)Dd * TWO_D;       // W_ch
  const size_t n3 = n2 + (size_t)Dd * TWO_D;       // W_cc
  const size_t n4 = n3 + (size_t)G8D;              // bias
  if (idx < n1)       WhhBf[idx]      = (__bf16)Whh[idx];
  else if (idx < n2)  WchBf[idx - n1] = (__bf16)Wch[idx - n1];
  else if (idx < n3)  WccBf[idx - n2] = (__bf16)Wcc[idx - n2];
  else if (idx < n4)  gbias[idx - n3] = bih[idx - n3] + bhh[idx - n3];
}

// ---------------------------------------------------------------------------
// Input projections: Ap[i][b][n] = emb[prem[b][i]] . W_ih[n][0:E]
//                    Bh[j][b][n] = emb[hyp [b][j]] . W_ih[n][E:2E]
// ---------------------------------------------------------------------------
__global__ void inproj_kernel(const int* __restrict__ premise,
                              const int* __restrict__ hypothesis,
                              const float* __restrict__ emb,
                              const float* __restrict__ Wih,
                              float* __restrict__ Ap,
                              float* __restrict__ Bh) {
  const size_t per = (size_t)Pp * Bb * G8D;
  size_t idx = (size_t)blockIdx.x * blockDim.x + threadIdx.x;
  if (idx >= 2 * per) return;
  int which = idx >= per;
  size_t r = which ? idx - per : idx;
  int n = (int)(r % G8D);
  size_t t = r / G8D;
  int b = (int)(t % Bb);
  int i = (int)(t / Bb);
  int tok = which ? hypothesis[b * HhLen + i] : premise[b * Pp + i];
  const float* ev = emb + (size_t)tok * Ee;
  const float* w  = Wih + (size_t)n * (2 * Ee) + (which ? Ee : 0);
  float s = 0.0f;
  for (int e = 0; e < Ee; ++e) s += ev[e] * w[e];
  (which ? Bh : Ap)[r] = s;
}

// ---------------------------------------------------------------------------
// Gates kernel: one wavefront (anti-diagonal d). 4 WGs per cell, 512 thr/WG.
// Each wave owns 16 columns of all 4 gates; computes preact via WMMA over
// hcat = [left_h, lower_h] (bf16), then LSTM elementwise -> h2,c2 (bf16).
// ---------------------------------------------------------------------------
__device__ __forceinline__ void gemm_half(const __bf16* __restrict__ hsrc,
                                          const __bf16* __restrict__ Whh,
                                          int l, int hi, int n0, int kwoff,
                                          v8f acc[2][4]) {
  // hsrc: [32][512] bf16 (one neighbor's h); Whh k-offset = kwoff + kk.
  for (int kk = 0; kk < Dd; kk += 32) {
    // prefetch next K-block of the weight stream (global_prefetch_b8)
    if (kk + 32 < Dd)
      __builtin_prefetch(Whh + (size_t)(n0 + l) * TWO_D + kwoff + kk + 32, 0, 1);
    v16bf a0 = load_frag(hsrc, l,      Dd, kk, hi);
    v16bf a1 = load_frag(hsrc, 16 + l, Dd, kk, hi);
    v16bf bf[4];
#pragma unroll
    for (int g = 0; g < 4; ++g)
      bf[g] = load_frag(Whh, g * TWO_D + n0 + l, TWO_D, kwoff + kk, hi);
#pragma unroll
    for (int g = 0; g < 4; ++g) {
      acc[0][g] = __builtin_amdgcn_wmma_f32_16x16x32_bf16(
          false, a0, false, bf[g], (short)0, acc[0][g], false, false);
      acc[1][g] = __builtin_amdgcn_wmma_f32_16x16x32_bf16(
          false, a1, false, bf[g], (short)0, acc[1][g], false, false);
    }
  }
}

__global__ __launch_bounds__(512) void gates_kernel(
    const float* __restrict__ Ap, const float* __restrict__ Bh,
    const float* __restrict__ gbias, const __bf16* __restrict__ Whh,
    const __bf16* __restrict__ HprevBf, const float* __restrict__ CprevF,
    __bf16* __restrict__ H2, __bf16* __restrict__ C2,
    int d, int i_min) {
  int cell = blockIdx.x >> 2;
  int part = blockIdx.x & 3;           // 256-column slab of each gate
  int i = i_min + cell;
  int j = d - i;
  int wave = (int)(threadIdx.x >> 5);
  int lane = (int)(threadIdx.x & 31);
  int l = lane & 15, hi = lane >> 4;
  int n0 = part * 256 + wave * 16;     // column base within each gate (=kcol base)

  const __bf16* leftH = (i > 0) ? (HprevBf + (size_t)j * Bb * Dd) : nullptr;
  const __bf16* lowH  = (j > 0) ? (HprevBf + (size_t)(j - 1) * Bb * Dd) : nullptr;
  const float*  leftC = (i > 0) ? (CprevF + (size_t)j * Bb * Dd) : nullptr;
  const float*  lowC  = (j > 0) ? (CprevF + (size_t)(j - 1) * Bb * Dd) : nullptr;

  const float* ap = Ap + (size_t)i * Bb * G8D;
  const float* bh = Bh + (size_t)j * Bb * G8D;

  // init accumulators with precomputed input projection + fused bias
  v8f acc[2][4];
#pragma unroll
  for (int m = 0; m < 2; ++m)
#pragma unroll
    for (int g = 0; g < 4; ++g) {
      int n = g * TWO_D + n0 + l;
      float bias = gbias[n];
#pragma unroll
      for (int v = 0; v < 8; ++v) {
        int bb = m * 16 + v + hi * 8;
        acc[m][g][v] = ap[(size_t)bb * G8D + n] + bh[(size_t)bb * G8D + n] + bias;
      }
    }

  // hcat GEMM: cols 0..511 = left_h, 512..1023 = lower_h (uniform guards)
  if (leftH) gemm_half(leftH, Whh, l, hi, n0, 0,  acc);
  if (lowH)  gemm_half(lowH,  Whh, l, hi, n0, Dd, acc);

  // LSTM elementwise over columns kcol = n0 + l (kcol side is wave-uniform:
  // n0 is a multiple of 16, so kcol < 512 iff n0 < 512)
  int kcol = n0 + l;
  const float* cSrc = (n0 < Dd) ? leftC : lowC;
  int ccol = (n0 < Dd) ? kcol : (kcol - Dd);
#pragma unroll
  for (int m = 0; m < 2; ++m) {
#pragma unroll
    for (int v = 0; v < 8; ++v) {
      int bb = m * 16 + v + hi * 8;
      float ig = sigm(acc[m][0][v]);
      float fg = sigm(acc[m][1][v]);
      float gg = tanhf(acc[m][2][v]);
      float og = sigm(acc[m][3][v]);
      float cp = cSrc ? cSrc[(size_t)bb * Dd + ccol] : 0.0f;
      float c2 = fg * cp + ig * gg;
      float h2 = og * tanhf(c2);
      size_t o = ((size_t)cell * Bb + bb) * TWO_D + kcol;
      H2[o] = (__bf16)h2;
      C2[o] = (__bf16)c2;
    }
  }
}

// ---------------------------------------------------------------------------
// Condense kernel: nh = h2@W_chT + b_ch ; nc = c2@W_ccT + b_cc  ([32,512]).
// Writes f32 state (for elementwise c-path + final MLP) and bf16 h (for the
// next wavefront's gate GEMM A-operand).
// ---------------------------------------------------------------------------
__global__ __launch_bounds__(512) void condense_kernel(
    const __bf16* __restrict__ H2, const __bf16* __restrict__ C2,
    const __bf16* __restrict__ Wch, const __bf16* __restrict__ Wcc,
    const float* __restrict__ bch, const float* __restrict__ bcc,
    float* __restrict__ HcurF, __bf16* __restrict__ HcurBf,
    float* __restrict__ CcurF, int d, int i_min) {
  int cell = blockIdx.x >> 1;
  int part = blockIdx.x & 1;
  int i = i_min + cell;
  int j = d - i;
  int wave = (int)(threadIdx.x >> 5);
  int lane = (int)(threadIdx.x & 31);
  int l = lane & 15, hi = lane >> 4;
  int n0 = part * 256 + wave * 16;

  const __bf16* h2 = H2 + (size_t)cell * Bb * TWO_D;
  const __bf16* c2 = C2 + (size_t)cell * Bb * TWO_D;

  v8f zero = {0.f, 0.f, 0.f, 0.f, 0.f, 0.f, 0.f, 0.f};
  v8f accH[2] = {zero, zero};
  v8f accC[2] = {zero, zero};

  for (int k0 = 0; k0 < TWO_D; k0 += 32) {
    if (k0 + 32 < TWO_D) {
      __builtin_prefetch(Wch + (size_t)(n0 + l) * TWO_D + k0 + 32, 0, 1);
      __builtin_prefetch(Wcc + (size_t)(n0 + l) * TWO_D + k0 + 32, 0, 1);
    }
    v16bf aH0 = load_frag(h2, l,      TWO_D, k0, hi);
    v16bf aH1 = load_frag(h2, 16 + l, TWO_D, k0, hi);
    v16bf aC0 = load_frag(c2, l,      TWO_D, k0, hi);
    v16bf aC1 = load_frag(c2, 16 + l, TWO_D, k0, hi);
    v16bf bH  = load_frag(Wch, n0 + l, TWO_D, k0, hi);
    v16bf bC  = load_frag(Wcc, n0 + l, TWO_D, k0, hi);
    accH[0] = __builtin_amdgcn_wmma_f32_16x16x32_bf16(false, aH0, false, bH,
                                                      (short)0, accH[0], false, false);
    accH[1] = __builtin_amdgcn_wmma_f32_16x16x32_bf16(false, aH1, false, bH,
                                                      (short)0, accH[1], false, false);
    accC[0] = __builtin_amdgcn_wmma_f32_16x16x32_bf16(false, aC0, false, bC,
                                                      (short)0, accC[0], false, false);
    accC[1] = __builtin_amdgcn_wmma_f32_16x16x32_bf16(false, aC1, false, bC,
                                                      (short)0, accC[1], false, false);
  }

  int n = n0 + l;
  float biasH = bch[n], biasC = bcc[n];
#pragma unroll
  for (int m = 0; m < 2; ++m) {
#pragma unroll
    for (int v = 0; v < 8; ++v) {
      int bb = m * 16 + v + hi * 8;
      float nh = accH[m][v] + biasH;
      float nc = accC[m][v] + biasC;
      size_t o = ((size_t)j * Bb + bb) * Dd + n;
      HcurF[o]  = nh;
      HcurBf[o] = (__bf16)nh;
      CcurF[o]  = nc;
    }
  }
}

// ---------------------------------------------------------------------------
// Final MLP + softmax (tiny, single WG): relu(h@W1T+b1) -> relu(@W2T+b2)
// -> softmax(@W3T+b3). h = condensed h of cell (P-1,H-1) (j = 23).
// ---------------------------------------------------------------------------
__global__ __launch_bounds__(256) void mlp_kernel(
    const float* __restrict__ Hfin,  // [24][32][512], use j=23
    const float* __restrict__ W1, const float* __restrict__ b1,
    const float* __restrict__ W2, const float* __restrict__ b2,
    const float* __restrict__ W3, const float* __restrict__ b3,
    float* __restrict__ t1, float* __restrict__ t2, float* __restrict__ out) {
  const float* h = Hfin + (size_t)(HhLen - 1) * Bb * Dd;
  int tid = (int)threadIdx.x;

  for (int idx = tid; idx < Bb * Dd; idx += 256) {
    int b = idx >> 9, n = idx & 511;
    float s = b1[n];
    for (int k = 0; k < Dd; ++k) s += h[(size_t)b * Dd + k] * W1[(size_t)n * Dd + k];
    t1[idx] = fmaxf(s, 0.0f);
  }
  __threadfence();
  __syncthreads();

  for (int idx = tid; idx < Bb * Dd; idx += 256) {
    int b = idx >> 9, n = idx & 511;
    float s = b2[n];
    for (int k = 0; k < Dd; ++k) s += t1[(size_t)b * Dd + k] * W2[(size_t)n * Dd + k];
    t2[idx] = fmaxf(s, 0.0f);
  }
  __threadfence();
  __syncthreads();

  if (tid < Bb) {
    int b = tid;
    float lg[3];
#pragma unroll
    for (int c = 0; c < 3; ++c) {
      float s = b3[c];
      for (int k = 0; k < Dd; ++k) s += t2[(size_t)b * Dd + k] * W3[(size_t)c * Dd + k];
      lg[c] = s;
    }
    float mx = fmaxf(lg[0], fmaxf(lg[1], lg[2]));
    float e0 = expf(lg[0] - mx), e1 = expf(lg[1] - mx), e2 = expf(lg[2] - mx);
    float sum = e0 + e1 + e2;
    out[b * 3 + 0] = e0 / sum;
    out[b * 3 + 1] = e1 / sum;
    out[b * 3 + 2] = e2 / sum;
  }
}

// ---------------------------------------------------------------------------
// Host-side orchestration
// ---------------------------------------------------------------------------
extern "C" void kernel_launch(void* const* d_in, const int* in_sizes, int n_in,
                              void* d_out, int out_size, void* d_ws, size_t ws_size,
                              hipStream_t stream) {
  (void)in_sizes; (void)n_in; (void)out_size; (void)ws_size;
  const int*   premise    = (const int*)  d_in[0];
  const int*   hypothesis = (const int*)  d_in[1];
  const float* emb        = (const float*)d_in[2];
  const float* W_ih       = (const float*)d_in[3];
  const float* W_hh       = (const float*)d_in[4];
  const float* b_ih       = (const float*)d_in[5];
  const float* b_hh       = (const float*)d_in[6];
  const float* W_ch       = (const float*)d_in[7];
  const float* b_ch       = (const float*)d_in[8];
  const float* W_cc       = (const float*)d_in[9];
  const float* b_cc       = (const float*)d_in[10];
  const float* W1 = (const float*)d_in[11];
  const float* b1 = (const float*)d_in[12];
  const float* W2 = (const float*)d_in[13];
  const float* b2 = (const float*)d_in[14];
  const float* W3 = (const float*)d_in[15];
  const float* b3 = (const float*)d_in[16];
  float* out = (float*)d_out;

  // ---- workspace carve-up (all offsets 256B-aligned)
  char* ws = (char*)d_ws;
  size_t off = 0;
  auto carve = [&](size_t bytes) -> char* {
    char* p = ws + off;
    off = (off + bytes + 255) & ~(size_t)255;
    return p;
  };
  __bf16* WhhBf = (__bf16*)carve((size_t)G8D * TWO_D * 2);
  __bf16* WchBf = (__bf16*)carve((size_t)Dd * TWO_D * 2);
  __bf16* WccBf = (__bf16*)carve((size_t)Dd * TWO_D * 2);
  float*  gbias = (float*) carve((size_t)G8D * 4);
  float*  Ap    = (float*) carve((size_t)Pp * Bb * G8D * 4);
  float*  Bh    = (float*) carve((size_t)HhLen * Bb * G8D * 4);
  __bf16* Hbf[2];
  float*  Hf[2];
  float*  Cf[2];
  for (int s = 0; s < 2; ++s) Hbf[s] = (__bf16*)carve((size_t)HhLen * Bb * Dd * 2);
  for (int s = 0; s < 2; ++s) Hf[s]  = (float*) carve((size_t)HhLen * Bb * Dd * 4);
  for (int s = 0; s < 2; ++s) Cf[s]  = (float*) carve((size_t)HhLen * Bb * Dd * 4);
  __bf16* H2 = (__bf16*)carve((size_t)HhLen * Bb * TWO_D * 2);
  __bf16* C2 = (__bf16*)carve((size_t)HhLen * Bb * TWO_D * 2);
  float*  t1 = (float*) carve((size_t)Bb * Dd * 4);
  float*  t2 = (float*) carve((size_t)Bb * Dd * 4);

  // ---- prep: bf16 weights + fused bias
  {
    size_t total = (size_t)G8D * TWO_D + 2 * (size_t)Dd * TWO_D + (size_t)G8D;
    int blocks = (int)((total + 255) / 256);
    prep_kernel<<<blocks, 256, 0, stream>>>(W_hh, W_ch, W_cc, b_ih, b_hh,
                                            WhhBf, WchBf, WccBf, gbias);
  }
  // ---- input projections (amortized across the whole grid)
  {
    size_t total = 2 * (size_t)Pp * Bb * G8D;
    int blocks = (int)((total + 255) / 256);
    inproj_kernel<<<blocks, 256, 0, stream>>>(premise, hypothesis, emb, W_ih, Ap, Bh);
  }

  // ---- anti-diagonal wavefront sweep
  const int ND = Pp + HhLen - 1;  // 47
  for (int d = 0; d < ND; ++d) {
    int i_min = (d > HhLen - 1) ? (d - (HhLen - 1)) : 0;
    int i_max = (d < Pp - 1) ? d : (Pp - 1);
    int ncells = i_max - i_min + 1;
    int rd = d & 1;        // read buffer index
    int wr = 1 - rd;       // write buffer index
    gates_kernel<<<ncells * 4, 512, 0, stream>>>(
        Ap, Bh, gbias, WhhBf, Hbf[rd], Cf[rd], H2, C2, d, i_min);
    condense_kernel<<<ncells * 2, 512, 0, stream>>>(
        H2, C2, WchBf, WccBf, b_ch, b_cc, Hf[wr], Hbf[wr], Cf[wr], d, i_min);
  }

  // last wavefront d=46 wrote into buffer index 1 - (46&1) = 1
  mlp_kernel<<<1, 256, 0, stream>>>(Hf[1], W1, b1, W2, b2, W3, b3, t1, t2, out);
}
